// RetrievalMemory_67954972557804
// MI455X (gfx1250) — matrix operand order repaired and checked by reading, not compiled, not run on hardware
//
#include <hip/hip_runtime.h>
#include <hip/hip_bf16.h>

// ---------------- problem constants (from reference) ----------------
#define HIDDEN 1024
#define MEMN   32
#define KEYD   32
#define VALD   32
#define TT     64
#define BB     2048
#define MTOT   (BB * TT)      // 131072 rows
#define NPAD   112            // 97 projection rows padded to 7 WMMA tiles
// workspace layout (bytes)
#define WS_WB_OFF   0                       // bf16 [NPAD][HIDDEN]  = 229376 B
#define WS_BIAS_OFF 229376                  // f32  [NPAD]          = 448 B
#define WS_P_OFF    230400                  // f32  [MTOT][NPAD]    = 58,720,256 B

typedef __attribute__((ext_vector_type(16))) __bf16 v16bf;
typedef __attribute__((ext_vector_type(8)))  float  v8f;

// ------------------------------------------------------------------
// Kernel 0: pack weights into padded bf16 matrix + bias vector.
// Row map: n in [0,32) -> W_query, [32,64) -> W_fk, [64,96) -> W_fv,
//          n == 96 -> W_gate, n in [97,112) -> zero.
// ------------------------------------------------------------------
__global__ void rm_prep_kernel(const float* __restrict__ Wg,  const float* __restrict__ bg,
                               const float* __restrict__ Wq,  const float* __restrict__ bq,
                               const float* __restrict__ Wfk, const float* __restrict__ bfk,
                               const float* __restrict__ Wfv, const float* __restrict__ bfv,
                               __bf16* __restrict__ Wb, float* __restrict__ biasArr) {
  int idx = blockIdx.x * blockDim.x + threadIdx.x;
  if (idx < NPAD * HIDDEN) {
    int n = idx / HIDDEN, k = idx - n * HIDDEN;
    float v = 0.f;
    if      (n < 32) v = Wq [n * HIDDEN + k];
    else if (n < 64) v = Wfk[(n - 32) * HIDDEN + k];
    else if (n < 96) v = Wfv[(n - 64) * HIDDEN + k];
    else if (n == 96) v = Wg[k];
    Wb[idx] = (__bf16)v;
  }
  if (idx < NPAD) {
    float v = 0.f;
    if      (idx < 32) v = bq [idx];
    else if (idx < 64) v = bfk[idx - 32];
    else if (idx < 96) v = bfv[idx - 64];
    else if (idx == 96) v = bg[0];
    biasArr[idx] = v;
  }
}

// ------------------------------------------------------------------
// Kernel 1: projection GEMM  P[M][112] = h[M][1024] @ Wb^T + bias
// Streams h once (512 MB -> memory bound, ~22 us at 23.3 TB/s).
// bf16 WMMA / f32 accumulate. WG = 256 threads = 8 waves; wave w owns
// M rows [blk*128 + w*16, +16) for all 7 N tiles; K chunked by 32.
//
// B tiles (112 rows x 32 k x bf16 = 7 KB) are staged into LDS with the
// CDNA5 async copy engine (global_load_async_to_lds_b128, ASYNCcnt),
// double-buffered: steady-state loop body issues the copy for kc+32,
// waits asynccnt<=2 (its own 2 in-order copies for kc are then done),
// barriers, computes kc from LDS, barriers. Last chunk peeled so the
// body is branch-free. All 7 B fragments are loaded into independent
// registers before the 7-WMMA chain so the ds_loads batch under one
// s_wait_dscnt instead of serializing per WMMA.
// ------------------------------------------------------------------
__global__ __launch_bounds__(256) void rm_proj_gemm_kernel(
    const float*  __restrict__ h,
    const __bf16* __restrict__ Wb,
    const float*  __restrict__ biasArr,
    float*        __restrict__ P) {
  __shared__ __align__(32) char ldsB[2 * 8192];   // double-buffered B tile

  const int tid   = threadIdx.x;
  const int lane  = tid & 31;
  const int wave  = tid >> 5;
  const int khalf = lane >> 4;
  const int l16   = lane & 15;
  const int Mbase = blockIdx.x * 128 + wave * 16;
  const int m     = Mbase + l16;

  const char* WbB = (const char*)Wb;
  // flat->LDS: low 32 bits of the flat shared pointer are the LDS byte addr
  const uint32_t ldsBase = (uint32_t)(uintptr_t)(void*)ldsB;

  // chunk ids for this thread's two async copies (16B granules)
  const uint32_t c0  = (uint32_t)tid;
  const uint32_t c1  = (uint32_t)tid + 256u;
  const uint32_t cc1 = (c1 < 448u) ? c1 : 447u;   // clamp source; dest stays unique

  v8f acc[7];
#pragma unroll
  for (int i = 0; i < 7; ++i) acc[i] = (v8f){0.f,0.f,0.f,0.f,0.f,0.f,0.f,0.f};

  const float* hrow = h + (size_t)m * HIDDEN;

  auto issue_btile_copy = [&](int bufi, int kci) {
    uint64_t g0 = (uint64_t)(uintptr_t)(WbB + (size_t)(c0 >> 2) * 2048 +
                                        (size_t)(c0 & 3u) * 16 + (size_t)kci * 2);
    uint64_t g1 = (uint64_t)(uintptr_t)(WbB + (size_t)(cc1 >> 2) * 2048 +
                                        (size_t)(cc1 & 3u) * 16 + (size_t)kci * 2);
    uint32_t l0 = ldsBase + (uint32_t)bufi * 8192u + c0 * 16u;
    uint32_t l1 = ldsBase + (uint32_t)bufi * 8192u + c1 * 16u;
    asm volatile("global_load_async_to_lds_b128 %0, %1, off"
                 :: "v"(l0), "v"(g0) : "memory");
    asm volatile("global_load_async_to_lds_b128 %0, %1, off"
                 :: "v"(l1), "v"(g1) : "memory");
  };

  auto compute_chunk = [&](int kc, int buf) {
    // prefetch the A stream ahead (speculative; OOB dropped by HW)
    __builtin_prefetch(hrow + kc + 256, 0, 3);

    const float* hp = hrow + kc + khalf * 16;
    float4 x0 = *(const float4*)(hp);        // K = base+0..3
    float4 x1 = *(const float4*)(hp + 4);    // K = base+4..7
    float4 y0 = *(const float4*)(hp + 8);    // K = base+8..11
    float4 y1 = *(const float4*)(hp + 12);   // K = base+12..15

    // batch all 7 B-fragment LDS reads into independent registers first
    const char* bufp = ldsB + buf * 8192;
    v16bf bf[7];
#pragma unroll
    for (int nt = 0; nt < 7; ++nt)
      bf[nt] = *(const v16bf*)(bufp + (nt * 16 + l16) * 64 + khalf * 32);

    v16bf a;
    a[0]  = (__bf16)x0.x; a[1]  = (__bf16)x0.y; a[2]  = (__bf16)x0.z; a[3]  = (__bf16)x0.w;
    a[4]  = (__bf16)x1.x; a[5]  = (__bf16)x1.y; a[6]  = (__bf16)x1.z; a[7]  = (__bf16)x1.w;
    a[8]  = (__bf16)y0.x; a[9]  = (__bf16)y0.y; a[10] = (__bf16)y0.z; a[11] = (__bf16)y0.w;
    a[12] = (__bf16)y1.x; a[13] = (__bf16)y1.y; a[14] = (__bf16)y1.z; a[15] = (__bf16)y1.w;

#pragma unroll
    for (int nt = 0; nt < 7; ++nt) {
      acc[nt] = __builtin_amdgcn_wmma_f32_16x16x32_bf16(
          /*neg_a=*/false, a, /*neg_b=*/false, bf[nt],
          /*c_mod=*/(short)0, acc[nt], /*reuse_a=*/false, /*reuse_b=*/false);
    }
  };

  issue_btile_copy(0, 0);

  // steady state: branch-free double-buffered pipeline
  for (int kc = 0; kc < HIDDEN - 32; kc += 32) {
    const int buf = (kc >> 5) & 1;
    issue_btile_copy(buf ^ 1, kc + 32);
    asm volatile("s_wait_asynccnt 0x2" ::: "memory");   // chunk kc landed
    __syncthreads();   // B1: whole WG's copies of buf are visible
    compute_chunk(kc, buf);
    __syncthreads();   // B2: reads done before buf is rewritten
  }
  // peeled last chunk
  asm volatile("s_wait_asynccnt 0x0" ::: "memory");
  __syncthreads();
  compute_chunk(HIDDEN - 32, ((HIDDEN - 32) >> 5) & 1);

  // Epilogue: C layout — lanes 0-15: (M=r, N=lane); lanes 16-31: (M=r+8, N=lane-16)
#pragma unroll
  for (int nt = 0; nt < 7; ++nt) {
    float bv = biasArr[nt * 16 + l16];
#pragma unroll
    for (int r = 0; r < 8; ++r) {
      int ml = r + khalf * 8;
      P[(size_t)(Mbase + ml) * NPAD + nt * 16 + l16] = acc[nt][r] + bv;
    }
  }
}

// ------------------------------------------------------------------
// Kernel 2: recurrent scan. One wave per batch element.
// Lane m holds K[m][0..31] (row layout) and Tf[m], mask[m].
// Lane v holds V[0..31][v] (column layout).
// q/fk are wave-uniform (same address across lanes -> broadcast loads).
// ------------------------------------------------------------------
__global__ __launch_bounds__(256) void rm_scan_kernel(
    const float* __restrict__ P,
    float* __restrict__ es,   // [B][T][32]
    float* __restrict__ ps) { // [B][T]
  const int lane = threadIdx.x & 31;
  const int b    = blockIdx.x * 8 + (threadIdx.x >> 5);

  float Krow[32], Vcol[32];
#pragma unroll
  for (int i = 0; i < 32; ++i) { Krow[i] = 0.f; Vcol[i] = 0.f; }
  float Tf = 0.f, mask = 0.f;
  const float scale = 0.17677669529663687f;  // 1/sqrt(32)

  for (int t = 0; t < TT; ++t) {
    const size_t row = (size_t)b * TT + t;
    const float* Pr = P + row * NPAD;

    float qarr[32], fkarr[32];
#pragma unroll
    for (int i = 0; i < 8; ++i) {
      float4 v4 = *(const float4*)(Pr + 4 * i);        // q (uniform)
      qarr[4*i+0] = v4.x; qarr[4*i+1] = v4.y; qarr[4*i+2] = v4.z; qarr[4*i+3] = v4.w;
      float4 f4 = *(const float4*)(Pr + 32 + 4 * i);   // fk (uniform)
      fkarr[4*i+0] = f4.x; fkarr[4*i+1] = f4.y; fkarr[4*i+2] = f4.z; fkarr[4*i+3] = f4.w;
    }
    float fvv = Pr[64 + lane];        // fv[lane]  (per-lane, coalesced)
    float g0  = Pr[96];               // gate logit (uniform)
    float p_reuse = 1.f / (1.f + __expf(-g0));

    // logits[m] = (K[m] . q) * scale     (lane = m)
    float lg = 0.f;
#pragma unroll
    for (int k = 0; k < 32; ++k) lg = fmaf(Krow[k], qarr[k], lg);
    lg *= scale;
    float ml = lg + (1.f - mask) * -1e9f;

    // masked softmax across lanes
    float mx = ml;
#pragma unroll
    for (int off = 16; off > 0; off >>= 1) mx = fmaxf(mx, __shfl_xor(mx, off));
    float ex = __expf(ml - mx);
    float sm = ex;
#pragma unroll
    for (int off = 16; off > 0; off >>= 1) sm += __shfl_xor(sm, off);
    float attn = ex / sm;

    // argmax(attn) (first max wins)
    float av = attn; int ai = lane;
#pragma unroll
    for (int off = 16; off > 0; off >>= 1) {
      float ov = __shfl_xor(av, off); int oi = __shfl_xor(ai, off);
      if (ov > av || (ov == av && oi < ai)) { av = ov; ai = oi; }
    }

    // empty_row / warmup / gate  (all wave-uniform)
    float msum = mask;
#pragma unroll
    for (int off = 16; off > 0; off >>= 1) msum += __shfl_xor(msum, off);
    float empty = (msum == 0.f) ? 1.f : 0.f;
    float warm  = (t < 4) ? 1.f : 0.f;
    float gate  = p_reuse * (1.f - empty) * (1.f - warm);

    // retrv[v] = sum_m attn[m] * V[m][v]   (lane = v; readlane broadcast)
    float rv = 0.f;
#pragma unroll
    for (int mm = 0; mm < 32; ++mm) rv = fmaf(__shfl(attn, mm), Vcol[mm], rv);

    float et = gate * rv + (1.f - gate) * fvv;
    es[row * 32 + lane] = et;
    if (lane == 0) ps[row] = p_reuse;

    // write decision + eviction (argmin of Tf + mask*1e-3, first min wins)
    float wr = (gate < 0.5f || warm > 0.f) ? 1.f : 0.f;
    float sv = Tf + mask * 0.001f; int si = lane;
#pragma unroll
    for (int off = 16; off > 0; off >>= 1) {
      float ov = __shfl_xor(sv, off); int oi = __shfl_xor(si, off);
      if (ov < sv || (ov == sv && oi < si)) { sv = ov; si = oi; }
    }
    const int e = si;                       // uniform after full butterfly
    const bool doWrite = (wr != 0.f);
    const bool mineRow = doWrite && (lane == e);
#pragma unroll
    for (int k = 0; k < 32; ++k) Krow[k] = mineRow ? fkarr[k] : Krow[k];
#pragma unroll
    for (int mm = 0; mm < 32; ++mm)
      Vcol[mm] = (doWrite && (mm == e)) ? fvv : Vcol[mm];

    float idx_oh   = (lane == e)  ? 1.f : 0.f;
    float reuse_oh = (lane == ai) ? 1.f : 0.f;
    float bump = wr * idx_oh + (1.f - wr) * reuse_oh;
    float Td = Tf * 0.85f;
    Tf = Td + bump * (1.f - Td);
    mask = fminf(mask + wr * idx_oh, 1.f);
  }
}

// ------------------------------------------------------------------
extern "C" void kernel_launch(void* const* d_in, const int* in_sizes, int n_in,
                              void* d_out, int out_size, void* d_ws, size_t ws_size,
                              hipStream_t stream) {
  const float* h   = (const float*)d_in[0];
  const float* Wg  = (const float*)d_in[1];
  const float* bg  = (const float*)d_in[2];
  const float* Wq  = (const float*)d_in[3];
  const float* bq  = (const float*)d_in[4];
  const float* Wfk = (const float*)d_in[5];
  const float* bfk = (const float*)d_in[6];
  const float* Wfv = (const float*)d_in[7];
  const float* bfv = (const float*)d_in[8];

  char*   ws      = (char*)d_ws;
  __bf16* Wb      = (__bf16*)(ws + WS_WB_OFF);
  float*  biasArr = (float*)(ws + WS_BIAS_OFF);
  float*  P       = (float*)(ws + WS_P_OFF);   // needs ~59 MB of workspace

  float* es = (float*)d_out;
  float* ps = es + (size_t)BB * TT * VALD;

  rm_prep_kernel<<<(NPAD * HIDDEN + 255) / 256, 256, 0, stream>>>(
      Wg, bg, Wq, bq, Wfk, bfk, Wfv, bfv, Wb, biasArr);
  rm_proj_gemm_kernel<<<MTOT / 128, 256, 0, stream>>>(h, Wb, biasArr, P);
  rm_scan_kernel<<<BB / 8, 256, 0, stream>>>(P, es, ps);
}
